// T5Encoder_89395449299779
// MI455X (gfx1250) — compile-verified, matrix operand order checked by reading, not run on hardware
//
#include <hip/hip_runtime.h>

typedef __attribute__((ext_vector_type(16))) _Float16 v16h;
typedef __attribute__((ext_vector_type(8)))  float    v8f;

#define NTOK   16384     // B*S = 8*2048
#define SEQ    2048
#define DMODEL 40
#define NPAD   48        // padded output-column dim
#define DPAD   64        // padded contraction dim (f16 rows)
#define DIN    128

__device__ __forceinline__ v8f wmma16(v16h a, v16h b, v8f c) {
  return __builtin_amdgcn_wmma_f32_16x16x32_f16(false, a, false, b, (short)0, c, false, false);
}

// A/B fragment for 16-bit 16x32 (or 32x16) tiles: lane_row points at this
// lane's row (A: M = lane&15 row of A; B: N = lane&15 column of B, with the
// source stored so that K is contiguous). half = lane>>4.
// a[2j+e] holds K = (j>=4?16:0) + (j&3)*2 + half*8 + e  ->  two contiguous
// 8-element (16-byte) groups per lane: [k_off + g*16 + half*8 .. +7], g=0,1.
__device__ __forceinline__ v16h load_frag(const _Float16* lane_row, int half, int k_off) {
  v16h a;
#pragma unroll
  for (int g = 0; g < 2; ++g) {
    int k0 = k_off + g * 16 + half * 8;
#pragma unroll
    for (int t = 0; t < 8; ++t) a[g * 8 + t] = lane_row[k0 + t];
  }
  return a;
}

__device__ __forceinline__ v16h load_afrag_f32(const float* lane_row, int half, int k_off) {
  v16h a;
#pragma unroll
  for (int g = 0; g < 2; ++g) {
    int k0 = k_off + g * 16 + half * 8;
#pragma unroll
    for (int t = 0; t < 8; ++t) a[g * 8 + t] = (_Float16)lane_row[k0 + t];
  }
  return a;
}

// ------------- weight prep: W[k][n] f32 -> Wt[n][k] f16, zero padded --------
__global__ void k_prep_w(const float* __restrict__ W, _Float16* __restrict__ Wt,
                         int Krows, int Ncols, int KPAD) {
  int idx = blockIdx.x * blockDim.x + threadIdx.x;
  int n = idx / KPAD, k = idx % KPAD;
  if (n >= NPAD) return;
  float v = (k < Krows && n < Ncols) ? W[k * Ncols + n] : 0.0f;
  Wt[(size_t)n * KPAD + k] = (_Float16)v;
}

// ---------------- bias table: pos_bias(rel) for rel in [-2047, 2047] --------
__global__ void k_bias_table(const float* __restrict__ rel_bias, float* __restrict__ tab) {
  int i = blockIdx.x * blockDim.x + threadIdx.x;
  if (i >= 4096) return;
  int rel = i - 2047;              // rel = key_pos - query_pos
  int b = (rel > 0) ? 16 : 0;      // num_buckets/2 = 16
  int r = rel < 0 ? -rel : rel;
  int bucket;
  if (r < 8) {
    bucket = b + r;
  } else {
    int large = 8 + (int)(__logf((float)r * 0.125f) / __logf(16.0f) * 8.0f);
    if (large > 15) large = 15;
    bucket = b + large;
  }
  tab[i] = rel_bias[bucket];
}

// ---------------- adapter: x = relu(embs @ Wa + ba) ------------------------
__global__ __launch_bounds__(128) void k_adapter(const float* __restrict__ embs,
                                                 const _Float16* __restrict__ WaT,
                                                 const float* __restrict__ ba,
                                                 float* __restrict__ x) {
  int lane = threadIdx.x & 31, wave = threadIdx.x >> 5;
  int tok0 = (blockIdx.x * 4 + wave) * 16;
  int half = lane >> 4, nl = lane & 15;
  const float* arow = embs + (size_t)(tok0 + nl) * DIN;
  v8f c[3] = { {}, {}, {} };
#pragma unroll
  for (int kc = 0; kc < 4; ++kc) {
    v16h a = load_afrag_f32(arow, half, kc * 32);
#pragma unroll
    for (int nt = 0; nt < 3; ++nt) {
      v16h b = load_frag(WaT + (size_t)(nt * 16 + nl) * DIN, half, kc * 32);
      c[nt] = wmma16(a, b, c[nt]);
    }
  }
#pragma unroll
  for (int nt = 0; nt < 3; ++nt) {
    int col = nt * 16 + nl;
    if (col < DMODEL) {
      float bias = ba[col];
#pragma unroll
      for (int r = 0; r < 8; ++r) {
        float v = c[nt][r] + bias;
        x[(size_t)(tok0 + r + half * 8) * DMODEL + col] = fmaxf(v, 0.0f);
      }
    }
  }
}

// ---------------- RMS norm -> f16 padded [tok][64] -------------------------
__global__ void k_rms16(const float* __restrict__ x, const float* __restrict__ g,
                        _Float16* __restrict__ h) {
  int tok = blockIdx.x * blockDim.x + threadIdx.x;
  if (tok >= NTOK) return;
  const float* row = x + (size_t)tok * DMODEL;
  float s = 0.0f;
  for (int d = 0; d < DMODEL; ++d) { float v = row[d]; s += v * v; }
  float inv = rsqrtf(s * (1.0f / DMODEL) + 1e-6f);
  _Float16* out = h + (size_t)tok * DPAD;
  for (int d = 0; d < DMODEL; ++d) out[d] = (_Float16)(row[d] * inv * g[d]);
  for (int d = DMODEL; d < DPAD; ++d) out[d] = (_Float16)0.0f;
}

// ---------------- QKV projections (V stored transposed) --------------------
__global__ __launch_bounds__(128) void k_qkv(const _Float16* __restrict__ h,
                                             const _Float16* __restrict__ WqT,
                                             const _Float16* __restrict__ WkT,
                                             const _Float16* __restrict__ WvT,
                                             _Float16* __restrict__ q16,
                                             _Float16* __restrict__ k16,
                                             _Float16* __restrict__ vT) {
  int lane = threadIdx.x & 31, wave = threadIdx.x >> 5;
  int tok0 = (blockIdx.x * 4 + wave) * 16;
  int half = lane >> 4, nl = lane & 15;
  const _Float16* arow = h + (size_t)(tok0 + nl) * DPAD;
  v16h a0 = load_frag(arow, half, 0);
  v16h a1 = load_frag(arow, half, 32);
  const _Float16* Ws[3] = { WqT, WkT, WvT };
  _Float16* outs[2] = { q16, k16 };
#pragma unroll
  for (int w = 0; w < 3; ++w) {
    v8f c[3] = { {}, {}, {} };
#pragma unroll
    for (int nt = 0; nt < 3; ++nt) {
      const _Float16* wrow = Ws[w] + (size_t)(nt * 16 + nl) * DPAD;
      c[nt] = wmma16(a0, load_frag(wrow, half, 0),  c[nt]);
      c[nt] = wmma16(a1, load_frag(wrow, half, 32), c[nt]);
    }
    if (w < 2) {
      _Float16* out = outs[w];
#pragma unroll
      for (int nt = 0; nt < 3; ++nt)
#pragma unroll
        for (int r = 0; r < 8; ++r)
          out[(size_t)(tok0 + r + half * 8) * DPAD + nt * 16 + nl] = (_Float16)c[nt][r];
#pragma unroll
      for (int r = 0; r < 8; ++r)
        out[(size_t)(tok0 + r + half * 8) * DPAD + 48 + nl] = (_Float16)0.0f;
    } else {
#pragma unroll
      for (int nt = 0; nt < 3; ++nt)
#pragma unroll
        for (int r = 0; r < 8; ++r)
          vT[(size_t)(nt * 16 + nl) * NTOK + tok0 + r + half * 8] = (_Float16)c[nt][r];
    }
  }
}

// ---------------- flash attention (1 head, no scaling, +pos bias) ----------
__global__ __launch_bounds__(256) void k_attn(const _Float16* __restrict__ q16,
                                              const _Float16* __restrict__ k16,
                                              const _Float16* __restrict__ vT,
                                              const float* __restrict__ tab,
                                              _Float16* __restrict__ ao) {
  __shared__ _Float16 Pb[8][16][32];     // per-wave P tile (C-layout -> A-layout)
  int lane = threadIdx.x & 31, wave = threadIdx.x >> 5;
  int qtile = blockIdx.x * 8 + wave;     // 0..1023
  int qb = qtile * 16;                   // first query token (global)
  int kbase = (qb >> 11) << 11;          // batch * SEQ
  int half = lane >> 4, nl = lane & 15;
  int qs = qb - kbase;                   // query seq position base

  const _Float16* qrow = q16 + (size_t)(qb + nl) * DPAD;
  v16h qa0 = load_frag(qrow, half, 0);
  v16h qa1 = load_frag(qrow, half, 32);

  float m[8], lsum[8];
  v8f o0 = {}, o1 = {}, o2 = {};
#pragma unroll
  for (int r = 0; r < 8; ++r) { m[r] = -1e30f; lsum[r] = 0.0f; }

  for (int it = 0; it < SEQ / 32; ++it) {
    int kb = kbase + it * 32;
    int ks = it * 32;
    const _Float16* kr0 = k16 + (size_t)(kb + nl) * DPAD;
    const _Float16* kr1 = k16 + (size_t)(kb + 16 + nl) * DPAD;

    // prefetch next tile's K / V rows while this tile computes
    __builtin_prefetch(kr0 + 32 * DPAD, 0, 1);
    __builtin_prefetch(kr1 + 32 * DPAD, 0, 1);
    __builtin_prefetch(vT + (size_t)nl * NTOK + kb + 32, 0, 1);

    v8f c0 = {}, c1 = {};
    c0 = wmma16(qa0, load_frag(kr0, half, 0),  c0);
    c0 = wmma16(qa1, load_frag(kr0, half, 32), c0);
    c1 = wmma16(qa0, load_frag(kr1, half, 0),  c1);
    c1 = wmma16(qa1, load_frag(kr1, half, 32), c1);

    // positional bias (Toeplitz table on rel = k - q)
#pragma unroll
    for (int r = 0; r < 8; ++r) {
      int rel0 = (ks + nl) - (qs + r + half * 8);
      c0[r] = c0[r] + tab[2047 + rel0];
      c1[r] = c1[r] + tab[2047 + rel0 + 16];
    }

    // online softmax
    float scale[8];
#pragma unroll
    for (int r = 0; r < 8; ++r) {
      float v = fmaxf(c0[r], c1[r]);
#pragma unroll
      for (int off = 1; off < 16; off <<= 1) v = fmaxf(v, __shfl_xor(v, off, 32));
      float mn = fmaxf(m[r], v);
      scale[r] = __expf(m[r] - mn);
      m[r] = mn;
    }
#pragma unroll
    for (int r = 0; r < 8; ++r) {
      float p0 = __expf(c0[r] - m[r]);
      float p1 = __expf(c1[r] - m[r]);
      float s = p0 + p1;
#pragma unroll
      for (int off = 1; off < 16; off <<= 1) s += __shfl_xor(s, off, 32);
      lsum[r] = lsum[r] * scale[r] + s;
      o0[r] = o0[r] * scale[r];
      o1[r] = o1[r] * scale[r];
      o2[r] = o2[r] * scale[r];
      int M = r + half * 8;
      Pb[wave][M][nl]      = (_Float16)p0;
      Pb[wave][M][nl + 16] = (_Float16)p1;
    }
    asm volatile("s_wait_dscnt 0" ::: "memory");   // intra-wave LDS RAW

    v16h pa = load_frag(&Pb[wave][nl][0], half, 0);   // P as A-fragment (16x32)
    o0 = wmma16(pa, load_frag(vT + (size_t)(nl)      * NTOK + kb, half, 0), o0);
    o1 = wmma16(pa, load_frag(vT + (size_t)(16 + nl) * NTOK + kb, half, 0), o1);
    o2 = wmma16(pa, load_frag(vT + (size_t)(32 + nl) * NTOK + kb, half, 0), o2);
  }

#pragma unroll
  for (int r = 0; r < 8; ++r) {
    float inv = 1.0f / lsum[r];
    size_t tok = (size_t)(qb + r + half * 8) * DPAD;
    ao[tok + nl]      = (_Float16)(o0[r] * inv);
    ao[tok + 16 + nl] = (_Float16)(o1[r] * inv);
    ao[tok + 32 + nl] = (_Float16)(o2[r] * inv);
    ao[tok + 48 + nl] = (_Float16)0.0f;
  }
}

// ---------------- "A(f16 padded) @ Wt + epilogue" kernels -------------------
__global__ __launch_bounds__(128) void k_oproj(const _Float16* __restrict__ ao,
                                               const _Float16* __restrict__ WoT,
                                               const float* __restrict__ x,
                                               float* __restrict__ x2) {
  int lane = threadIdx.x & 31, wave = threadIdx.x >> 5;
  int tok0 = (blockIdx.x * 4 + wave) * 16;
  int half = lane >> 4, nl = lane & 15;
  const _Float16* arow = ao + (size_t)(tok0 + nl) * DPAD;
  v16h a0 = load_frag(arow, half, 0);
  v16h a1 = load_frag(arow, half, 32);
  v8f c[3] = { {}, {}, {} };
#pragma unroll
  for (int nt = 0; nt < 3; ++nt) {
    const _Float16* wrow = WoT + (size_t)(nt * 16 + nl) * DPAD;
    c[nt] = wmma16(a0, load_frag(wrow, half, 0),  c[nt]);
    c[nt] = wmma16(a1, load_frag(wrow, half, 32), c[nt]);
  }
#pragma unroll
  for (int nt = 0; nt < 3; ++nt) {
    int col = nt * 16 + nl;
    if (col < DMODEL) {
#pragma unroll
      for (int r = 0; r < 8; ++r) {
        size_t idx = (size_t)(tok0 + r + half * 8) * DMODEL + col;
        x2[idx] = x[idx] + c[nt][r];
      }
    }
  }
}

__global__ __launch_bounds__(128) void k_ffn1(const _Float16* __restrict__ h2,
                                              const _Float16* __restrict__ wiT,
                                              _Float16* __restrict__ f) {
  int lane = threadIdx.x & 31, wave = threadIdx.x >> 5;
  int tok0 = (blockIdx.x * 4 + wave) * 16;
  int half = lane >> 4, nl = lane & 15;
  const _Float16* arow = h2 + (size_t)(tok0 + nl) * DPAD;
  v16h a0 = load_frag(arow, half, 0);
  v16h a1 = load_frag(arow, half, 32);
  v8f c[3] = { {}, {}, {} };
#pragma unroll
  for (int nt = 0; nt < 3; ++nt) {
    const _Float16* wrow = wiT + (size_t)(nt * 16 + nl) * DPAD;
    c[nt] = wmma16(a0, load_frag(wrow, half, 0),  c[nt]);
    c[nt] = wmma16(a1, load_frag(wrow, half, 32), c[nt]);
  }
#pragma unroll
  for (int nt = 0; nt < 3; ++nt)
#pragma unroll
    for (int r = 0; r < 8; ++r)
      f[(size_t)(tok0 + r + half * 8) * DPAD + nt * 16 + nl] =
          (_Float16)fmaxf(c[nt][r], 0.0f);
#pragma unroll
  for (int r = 0; r < 8; ++r)
    f[(size_t)(tok0 + r + half * 8) * DPAD + 48 + nl] = (_Float16)0.0f;
}

__global__ __launch_bounds__(128) void k_ffn2(const _Float16* __restrict__ f,
                                              const _Float16* __restrict__ woT,
                                              const float* __restrict__ x2,
                                              float* __restrict__ x3) {
  int lane = threadIdx.x & 31, wave = threadIdx.x >> 5;
  int tok0 = (blockIdx.x * 4 + wave) * 16;
  int half = lane >> 4, nl = lane & 15;
  const _Float16* arow = f + (size_t)(tok0 + nl) * DPAD;
  v16h a0 = load_frag(arow, half, 0);
  v16h a1 = load_frag(arow, half, 32);
  v8f c[3] = { {}, {}, {} };
#pragma unroll
  for (int nt = 0; nt < 3; ++nt) {
    const _Float16* wrow = woT + (size_t)(nt * 16 + nl) * DPAD;
    c[nt] = wmma16(a0, load_frag(wrow, half, 0),  c[nt]);
    c[nt] = wmma16(a1, load_frag(wrow, half, 32), c[nt]);
  }
#pragma unroll
  for (int nt = 0; nt < 3; ++nt) {
    int col = nt * 16 + nl;
    if (col < DMODEL) {
#pragma unroll
      for (int r = 0; r < 8; ++r) {
        size_t idx = (size_t)(tok0 + r + half * 8) * DMODEL + col;
        x3[idx] = x2[idx] + c[nt][r];
      }
    }
  }
}

// ---------------- final RMS norm -> f32 output ------------------------------
__global__ void k_rms_out(const float* __restrict__ x, const float* __restrict__ g,
                          float* __restrict__ out) {
  int tok = blockIdx.x * blockDim.x + threadIdx.x;
  if (tok >= NTOK) return;
  const float* row = x + (size_t)tok * DMODEL;
  float s = 0.0f;
  for (int d = 0; d < DMODEL; ++d) { float v = row[d]; s += v * v; }
  float inv = rsqrtf(s * (1.0f / DMODEL) + 1e-6f);
  float* o = out + (size_t)tok * DMODEL;
  for (int d = 0; d < DMODEL; ++d) o[d] = row[d] * inv * g[d];
}

extern "C" void kernel_launch(void* const* d_in, const int* in_sizes, int n_in,
                              void* d_out, int out_size, void* d_ws, size_t ws_size,
                              hipStream_t stream) {
  const float* embs     = (const float*)d_in[0];
  const float* Wa       = (const float*)d_in[1];
  const float* ba       = (const float*)d_in[2];
  const float* ln1      = (const float*)d_in[3];
  const float* Wq       = (const float*)d_in[4];
  const float* Wk       = (const float*)d_in[5];
  const float* Wv       = (const float*)d_in[6];
  const float* Wo       = (const float*)d_in[7];
  const float* ln2      = (const float*)d_in[8];
  const float* wi       = (const float*)d_in[9];
  const float* wo       = (const float*)d_in[10];
  const float* lnf      = (const float*)d_in[11];
  const float* rel_bias = (const float*)d_in[12];

  char* ws = (char*)d_ws;
  float*    tab  = (float*)ws;                               // 16 KB
  _Float16* WaT  = (_Float16*)(ws + 16384);                  // 48x128 f16
  _Float16* WqT  = (_Float16*)(ws + 16384 + 12288);          // 48x64 f16 each
  _Float16* WkT  = WqT + NPAD * DPAD;
  _Float16* WvT  = WkT + NPAD * DPAD;
  _Float16* WoT  = WvT + NPAD * DPAD;
  _Float16* wiT  = WoT + NPAD * DPAD;
  _Float16* woT  = wiT + NPAD * DPAD;
  char* dyn = ws + 65536;
  float*    x    = (float*)dyn;                              // [NTOK][40] f32
  float*    x2   = (float*)(dyn + 2621440);                  // [NTOK][40] f32
  _Float16* hao  = (_Float16*)(dyn + 2 * 2621440);           // h, then attn-out
  _Float16* qh2  = (_Float16*)((char*)hao + 2097152);        // q, then h2
  _Float16* kfb  = (_Float16*)((char*)qh2 + 2097152);        // k, then ffn act
  _Float16* vT   = (_Float16*)((char*)kfb + 2097152);        // [64][NTOK]

  k_bias_table<<<16, 256, 0, stream>>>(rel_bias, tab);
  k_prep_w<<<24, 256, 0, stream>>>(Wa, WaT, DIN, DMODEL, DIN);    // 48*128
  k_prep_w<<<12, 256, 0, stream>>>(Wq, WqT, DMODEL, DMODEL, DPAD); // 48*64
  k_prep_w<<<12, 256, 0, stream>>>(Wk, WkT, DMODEL, DMODEL, DPAD);
  k_prep_w<<<12, 256, 0, stream>>>(Wv, WvT, DMODEL, DMODEL, DPAD);
  k_prep_w<<<12, 256, 0, stream>>>(Wo, WoT, DMODEL, DMODEL, DPAD);
  k_prep_w<<<12, 256, 0, stream>>>(wi, wiT, DMODEL, DMODEL, DPAD);
  k_prep_w<<<12, 256, 0, stream>>>(wo, woT, DMODEL, DMODEL, DPAD);

  k_adapter  <<<256, 128, 0, stream>>>(embs, WaT, ba, x);
  k_rms16    <<<64, 256, 0, stream>>>(x, ln1, hao);
  k_qkv      <<<256, 128, 0, stream>>>(hao, WqT, WkT, WvT, qh2, kfb, vT);
  k_attn     <<<128, 256, 0, stream>>>(qh2, kfb, vT, tab, hao);
  k_oproj    <<<256, 128, 0, stream>>>(hao, WoT, x, x2);
  k_rms16    <<<64, 256, 0, stream>>>(x2, ln2, qh2);
  k_ffn1     <<<256, 128, 0, stream>>>(qh2, wiT, kfb);
  k_ffn2     <<<256, 128, 0, stream>>>(kfb, woT, x2, x);
  k_rms_out  <<<64, 256, 0, stream>>>(x, lnf, (float*)d_out);
}
